// ESNi_5394478924030
// MI455X (gfx1250) — compile-verified, matrix operand order checked by reading, not run on hardware
//
#include <hip/hip_runtime.h>
#include <stdint.h>

// ---- shapes -----------------------------------------------------------------
#define LEAK  0.8f
#define BB    4          // batch
#define TT    2048       // sequence length
#define VV    32000      // vocab
#define NN    2048       // reservoir width
#define SCAN_BLOCKS 32   // persistent scan blocks (must all be co-resident)

typedef __attribute__((ext_vector_type(16))) __bf16 v16bf;
typedef __attribute__((ext_vector_type(8)))  __bf16 v8bf;
typedef __attribute__((ext_vector_type(8)))  float  v8f;
typedef __attribute__((ext_vector_type(4)))  unsigned int u32x4;
typedef __attribute__((ext_vector_type(4)))  int i32x4;

#if defined(__AMDGCN__) && __has_builtin(__builtin_amdgcn_global_load_async_to_lds_b128)
#define HAVE_ASYNC 1
typedef __attribute__((address_space(1))) i32x4 as1_i32x4;
typedef __attribute__((address_space(3))) i32x4 as3_i32x4;
#else
#define HAVE_ASYNC 0
#endif

// float -> bf16, round-to-nearest-even
__device__ __forceinline__ unsigned short f2bf(float f) {
    unsigned int u = __float_as_uint(f);
    u += 0x7FFFu + ((u >> 16) & 1u);
    return (unsigned short)(u >> 16);
}

__device__ __forceinline__ v16bf ldbf16(const unsigned short* p) {
    return *reinterpret_cast<const v16bf*>(p);  // 32B aligned (global paths)
}

// LDS fragment load: rows are only 16B aligned (80B stride), so do 2x16B.
__device__ __forceinline__ v16bf ldlds(const unsigned short* p) {
    v8bf lo = *reinterpret_cast<const v8bf*>(p);
    v8bf hi = *reinterpret_cast<const v8bf*>(p + 8);
    return __builtin_shufflevector(lo, hi, 0,1,2,3,4,5,6,7,8,9,10,11,12,13,14,15);
}

// 16-byte global -> LDS copy: async on CDNA5 (ASYNCcnt), sync fallback otherwise.
__device__ __forceinline__ void cp16(void* lptr, const void* gptr) {
#if HAVE_ASYNC
    __builtin_amdgcn_global_load_async_to_lds_b128(
        (as1_i32x4*)gptr, (as3_i32x4*)lptr, 0, 0);
#else
    *reinterpret_cast<u32x4*>(lptr) = *reinterpret_cast<const u32x4*>(gptr);
#endif
}

__device__ __forceinline__ void wait_async() {
#if HAVE_ASYNC
#  if __has_builtin(__builtin_amdgcn_s_wait_asynccnt)
    __builtin_amdgcn_s_wait_asynccnt(0);
#  else
    asm volatile("s_wait_asynccnt 0x0" ::: "memory");
#  endif
#endif
}

// ---- conversion / init kernels ---------------------------------------------
// W_rec (f32, [k][n]) -> bf16 transposed [n][k] so WMMA-B loads are K-contiguous
__global__ void k_wrec_t(const float* __restrict__ wrec,
                         unsigned short* __restrict__ wt) {
    int idx = blockIdx.x * blockDim.x + threadIdx.x;
    if (idx >= NN * NN) return;
    int n = idx / NN;
    int k = idx - n * NN;
    wt[idx] = f2bf(wrec[k * NN + n]);
}

// W_out (f32, [v][k]) -> bf16 same layout (already K-contiguous per vocab row)
__global__ void k_wout(const float* __restrict__ w,
                       unsigned short* __restrict__ o, int n) {
    int idx = blockIdx.x * blockDim.x + threadIdx.x;
    if (idx < n) o[idx] = f2bf(w[idx]);
}

// zero h ping-pong buffers (f32 state + bf16 mirror, 16 padded rows) + barrier
__global__ void k_init(unsigned short* __restrict__ hbf,
                       float* __restrict__ hf,
                       int* __restrict__ bar) {
    int idx = blockIdx.x * blockDim.x + threadIdx.x;
    if (idx < 2 * 16 * NN) { hbf[idx] = 0; hf[idx] = 0.0f; }
    if (idx == 0) *bar = 0;
}

// ---- persistent recurrent scan ----------------------------------------------
// h_new = (1-a)h + a*relu(h@W_rec + emb[x[:,t]]), 2048 sequential steps.
// 32 blocks x 128 threads (4 waves); each wave owns one 16-column tile.
// Steps are separated by a device-scope atomic grid barrier; W_rec (8.4 MB
// bf16) stays L2-resident across the whole scan.
__global__ void __launch_bounds__(128) k_scan(
    const unsigned short* __restrict__ wrt,   // [NN][NN] bf16 (transposed)
    unsigned short*       __restrict__ hbf,   // [2][16][NN] bf16 ping-pong
    float*                __restrict__ hf,    // [2][16][NN] f32 ping-pong
    unsigned short*       __restrict__ hsb,   // [TT*BB][NN] bf16
    const float*          __restrict__ emb,   // [VV][NN] f32
    const int*            __restrict__ x,     // [BB][TT]
    int*                  __restrict__ bar)   // grid barrier counter
{
    const int wave = threadIdx.x >> 5;
    const int lane = threadIdx.x & 31;
    const int col0 = (blockIdx.x * 4 + wave) * 16;   // 32*4*16 = 2048 cols
    const int l15  = lane & 15;
    const int koff = (lane >> 4) << 4;               // K group per half-wave
    const int n     = col0 + l15;
    const int mbase = (lane >> 4) * 8;

    const unsigned short* bp = wrt + (size_t)(col0 + l15) * NN + koff;  // B: col=N

    for (int t = 0; t < TT; ++t) {
        const int pp = t & 1;
        const unsigned short* hb_prev = hbf + pp * 16 * NN;
        unsigned short*       hb_next = hbf + (pp ^ 1) * 16 * NN;
        const float*          hf_prev = hf + pp * 16 * NN;
        float*                hf_next = hf + (pp ^ 1) * 16 * NN;
        const unsigned short* ap = hb_prev + l15 * NN + koff;           // A: row=M

        v8f acc = {};
        #pragma unroll 4
        for (int k = 0; k < NN; k += 32) {
            v16bf a = ldbf16(ap + k);
            v16bf b = ldbf16(bp + k);
            acc = __builtin_amdgcn_wmma_f32_16x16x32_bf16(
                false, a, false, b, (short)0, acc, false, false);
        }

        // epilogue: C layout -> lane holds column n, VGPR j holds row mbase+j
        #pragma unroll
        for (int j = 0; j < 8; ++j) {
            int m = mbase + j;
            if (m < BB) {
                float u   = emb[(size_t)x[m * TT + t] * NN + n];
                float pre = acc[j] + u;
                float hn  = (1.0f - LEAK) * hf_prev[m * NN + n]
                          + LEAK * fmaxf(pre, 0.0f);
                hf_next[m * NN + n] = hn;
                unsigned short hb = f2bf(hn);
                hb_next[m * NN + n] = hb;
                hsb[(size_t)(t * BB + m) * NN + n] = hb;
            }
        }

        // ---- grid barrier: all 32 blocks must finish step t -----------------
        __threadfence();          // each wave flushes its h stores to device scope
        __syncthreads();
        if (threadIdx.x == 0) {
            atomicAdd(bar, 1);
            unsigned goal = (unsigned)(t + 1) * (unsigned)SCAN_BLOCKS;
            while ((unsigned)__hip_atomic_load(bar, __ATOMIC_ACQUIRE,
                                               __HIP_MEMORY_SCOPE_AGENT) < goal) {
                __builtin_amdgcn_s_sleep(1);
            }
        }
        __syncthreads();
        __threadfence();          // acquire: invalidate stale cached h lines
    }
}

// ---- output projection: logits = hs @ W_out^T + b_out -----------------------
// Block = 256 threads (8 waves as 4(M) x 2(N)) -> 128x128 C tile; wave = 32x64.
// K is staged through double-buffered LDS with async global->LDS copies
// overlapping WMMA compute. Grid = (32000/128, 8192/128) = (250, 64).
#define KSTEP   32
#define ROWPAD  40                 // 32 bf16 + 8 pad elements = 80 B row stride
#define STAGE_U (2 * 128 * ROWPAD) // A panel + B panel, in ushorts

__global__ void __launch_bounds__(256) k_proj(
    const unsigned short* __restrict__ hsb,   // [8192][NN] bf16
    const unsigned short* __restrict__ wob,   // [VV][NN] bf16
    const float*          __restrict__ bias,  // [VV]
    float*                __restrict__ out)   // [BB][TT][VV]
{
    __shared__ __attribute__((aligned(32))) unsigned short smem[2 * STAGE_U];

    const int tid  = threadIdx.x;
    const int wave = tid >> 5;
    const int lane = tid & 31;
    const int wm = wave & 3;          // 0..3  (M direction)
    const int wn = wave >> 2;         // 0..1  (N direction)
    const int mblk = blockIdx.y * 128;
    const int nblk = blockIdx.x * 128;

    const int l15  = lane & 15;
    const int koff = (lane >> 4) << 4;   // 0 or 16 elements

    // copy role: thread tid handles one 64B row-slab per stage
    const bool isA = tid < 128;
    const int  crow = isA ? tid : tid - 128;
    const unsigned short* gsrc =
        (isA ? hsb + (size_t)(mblk + crow) * NN
             : wob + (size_t)(nblk + crow) * NN);
    unsigned short* ldst_base = smem + (isA ? 0 : 128 * ROWPAD) + crow * ROWPAD;

    #define STAGE(s, k0)                                                     \
        do {                                                                 \
            const unsigned short* g = gsrc + (k0);                           \
            unsigned short* l = ldst_base + (s) * STAGE_U;                   \
            cp16(l + 0,  g + 0);                                             \
            cp16(l + 8,  g + 8);                                             \
            cp16(l + 16, g + 16);                                            \
            cp16(l + 24, g + 24);                                            \
        } while (0)

    v8f acc[2][4] = {};
    STAGE(0, 0);

    const int nsteps = NN / KSTEP;   // 64
    for (int ks = 0; ks < nsteps; ++ks) {
        const int cur = ks & 1;
        wait_async();
        __syncthreads();             // stage `cur` visible to all waves
        if (ks + 1 < nsteps) STAGE(cur ^ 1, (ks + 1) * KSTEP);

        const unsigned short* As = smem + cur * STAGE_U;
        const unsigned short* Bs = As + 128 * ROWPAD;

        v16bf af[2], bf[4];
        #pragma unroll
        for (int i = 0; i < 2; ++i)
            af[i] = ldlds(As + (wm * 32 + 16 * i + l15) * ROWPAD + koff);
        #pragma unroll
        for (int j = 0; j < 4; ++j)
            bf[j] = ldlds(Bs + (wn * 64 + 16 * j + l15) * ROWPAD + koff);

        #pragma unroll
        for (int i = 0; i < 2; ++i)
            #pragma unroll
            for (int j = 0; j < 4; ++j)
                acc[i][j] = __builtin_amdgcn_wmma_f32_16x16x32_bf16(
                    false, af[i], false, bf[j], (short)0, acc[i][j], false, false);

        __syncthreads();             // everyone done reading `cur` before reuse
    }

    // epilogue: row m = t*BB + b  ->  out[b][t][v] = acc + bias[v]
    const int mbase = (lane >> 4) * 8;
    #pragma unroll
    for (int i = 0; i < 2; ++i) {
        #pragma unroll
        for (int j2 = 0; j2 < 4; ++j2) {
            int nn = nblk + wn * 64 + j2 * 16 + l15;
            float bz = bias[nn];
            #pragma unroll
            for (int j = 0; j < 8; ++j) {
                int m = mblk + wm * 32 + i * 16 + mbase + j;
                int tt = m >> 2;
                int bb = m & 3;
                out[((size_t)bb * TT + tt) * (size_t)VV + nn] = acc[i][j2][j] + bz;
            }
        }
    }
}

// ---- launcher ---------------------------------------------------------------
extern "C" void kernel_launch(void* const* d_in, const int* in_sizes, int n_in,
                              void* d_out, int out_size, void* d_ws, size_t ws_size,
                              hipStream_t stream) {
    const int*   x    = (const int*)  d_in[0];
    const float* emb  = (const float*)d_in[1];
    const float* wrec = (const float*)d_in[2];
    const float* wout = (const float*)d_in[3];
    const float* bout = (const float*)d_in[4];
    float* out = (float*)d_out;

    // workspace layout (all 256B aligned):
    //   [0)            WrecT bf16   : 2048*2048*2   =   8,388,608
    //   [8388608)      Wout  bf16   : 32000*2048*2  = 131,072,000
    //   [139460608)    hs    bf16   : 8192*2048*2   =  33,554,432
    //   [173015040)    h f32 x2     : 2*16*2048*4   =     262,144
    //   [173277184)    h bf16 x2    : 2*16*2048*2   =     131,072
    //   [173408256)    barrier      : 4
    char* ws = (char*)d_ws;
    unsigned short* wrt = (unsigned short*)(ws);
    unsigned short* wob = (unsigned short*)(ws + 8388608);
    unsigned short* hsb = (unsigned short*)(ws + 139460608);
    float*          hf  = (float*)         (ws + 173015040);
    unsigned short* hbf = (unsigned short*)(ws + 173277184);
    int*            bar = (int*)           (ws + 173408256);

    k_wrec_t<<<(NN * NN + 255) / 256, 256, 0, stream>>>(wrec, wrt);
    k_wout  <<<(VV * NN + 255) / 256, 256, 0, stream>>>(wout, wob, VV * NN);
    k_init  <<<(2 * 16 * NN + 255) / 256, 256, 0, stream>>>(hbf, hf, bar);

    // persistent reservoir scan: one launch, grid barrier between timesteps
    k_scan<<<SCAN_BLOCKS, 128, 0, stream>>>(wrt, hbf, hf, hsb, emb, x, bar);

    // compute-bound bf16 WMMA projection (1.07 TFLOP), async-LDS pipelined
    k_proj<<<dim3(250, 64), 256, 0, stream>>>(hsb, wob, bout, out);
}
